// MixedFFN_60172491817772
// MI455X (gfx1250) — matrix operand order repaired and checked by reading, not compiled, not run on hardware
//
#include <hip/hip_runtime.h>
#include <stdint.h>

// ---------------------------------------------------------------------------
// MixedFFN for MI455X (gfx1250, wave32, WMMA bf16 16x16x32, f32 accumulate)
// B=8, T=2048, D=1024, F=2048, LNS=64, s = T-LNS = 1984
// Shared path: direct-from-global WMMA GEMMs (weights/acts fit in 192MB L2).
// NS path:     TDM (tensor_load_to_lds) double-buffered fp32 weight tiles.
// ---------------------------------------------------------------------------

typedef __attribute__((ext_vector_type(16))) __bf16 v16bf;
typedef __attribute__((ext_vector_type(8)))  float  v8f;
typedef unsigned short ushort_t;
typedef __attribute__((ext_vector_type(8)))  ushort_t v8us;
typedef __attribute__((ext_vector_type(16))) ushort_t v16us;
typedef __attribute__((ext_vector_type(4)))  unsigned int v4u;
typedef __attribute__((ext_vector_type(8)))  int v8i;
typedef __attribute__((ext_vector_type(4)))  int v4i;

static constexpr int B_  = 8;
static constexpr int T_  = 2048;
static constexpr int D_  = 1024;
static constexpr int F_  = 2048;
static constexpr int LNS = 64;
static constexpr int S_  = T_ - LNS;          // 1984 (multiple of 16)

__device__ __forceinline__ ushort_t f2bf(float f) {
  __bf16 h = (__bf16)f;                        // hardware cvt, RNE
  return __builtin_bit_cast(unsigned short, h);
}

__device__ __forceinline__ float gelu_exact(float v) {
  return 0.5f * v * (1.0f + erff(v * 0.70710678118654752f));
}

// ---- WMMA fragment helpers (contiguous k = 16*(lane>>4) + s in A and B) -----

// A 16x32 bf16 fragment. prow = &A[(m0 + (lane&15)) * lda + k0], ksel = lane>>4.
__device__ __forceinline__ v16bf frag_a(const ushort_t* prow, int ksel) {
  v8us lo = *(const v8us*)(prow + ksel * 16);
  v8us hi = *(const v8us*)(prow + ksel * 16 + 8);
  v16us t;
#pragma unroll
  for (int i = 0; i < 8; ++i) { t[i] = lo[i]; t[i + 8] = hi[i]; }
  return __builtin_bit_cast(v16bf, t);
}

// B 32x16 bf16 fragment from "N-major" weights (row n contiguous along K).
__device__ __forceinline__ v16bf frag_b_nmajor(const ushort_t* pn, int ksel) {
  v16us t = *(const v16us*)(pn + ksel * 16);
  return __builtin_bit_cast(v16bf, t);
}

__device__ __forceinline__ v8f wmma_bf16(v16bf a, v16bf b, v8f c) {
  return __builtin_amdgcn_wmma_f32_16x16x32_bf16(false, a, false, b,
                                                 (short)0, c, false, false);
}

// ---- Tensor Data Mover: 2D fp32 tile (tile_k rows x tile_n cols) -> LDS ---
// D# packing per CDNA5 ISA. 6-arg builtin form.
__device__ __forceinline__ void tdm_load_2d_f32(unsigned lds_byte_off,
                                                const float* gaddr,
                                                unsigned tile_n, unsigned tile_k,
                                                unsigned row_stride_elems) {
  unsigned long long ga = (unsigned long long)(uintptr_t)gaddr;
  v4u g0;
  g0[0] = 1u;                                   // count=1, user mode, no gather
  g0[1] = lds_byte_off;                         // lds_addr (bytes)
  g0[2] = (unsigned)(ga & 0xFFFFFFFFu);         // global_addr[31:0]
  g0[3] = (unsigned)((ga >> 32) & 0x01FFFFFFu)  // global_addr[56:32]
        | (2u << 30);                           // type = 2 ("image")
  const unsigned tensor_d0 = row_stride_elems;  // never OOB for our tiles
  const unsigned tensor_d1 = 0x100000u;
  v8i g1;
  g1[0] = (int)(2u << 16);                      // wg_mask=0, data_size=2 (4B)
  g1[1] = (int)((tensor_d0 & 0xFFFFu) << 16);   // abar=0 | tensor_dim0 lo16
  g1[2] = (int)(((tensor_d0 >> 16) & 0xFFFFu) | ((tensor_d1 & 0xFFFFu) << 16));
  g1[3] = (int)(((tensor_d1 >> 16) & 0xFFFFu) | (tile_n << 16));  // tile_dim0
  g1[4] = (int)(tile_k & 0xFFFFu);              // tile_dim1 | tile_dim2=0
  g1[5] = (int)row_stride_elems;                // tensor_dim0_stride[31:0]
  g1[6] = 0;
  g1[7] = 0;
  v4i z4; z4[0] = 0; z4[1] = 0; z4[2] = 0; z4[3] = 0;
  v8i z8;
#pragma unroll
  for (int i = 0; i < 8; ++i) z8[i] = 0;
  __builtin_amdgcn_tensor_load_to_lds(g0, g1, z4, z4, z8, 0);
}

// ---------------------------- convert kernels ------------------------------

__global__ void k_f32_to_bf16(const float* __restrict__ src,
                              ushort_t* __restrict__ dst, long long n) {
  long long stride = (long long)gridDim.x * blockDim.x;
  for (long long i = (long long)blockIdx.x * blockDim.x + threadIdx.x; i < n; i += stride)
    dst[i] = f2bf(src[i]);
}

// Pack NS activations: Xns[t][m][d], m=0..7 -> batch b=m at timestep s+t, m=8..15 -> 0
__global__ void k_build_xns(const float* __restrict__ x, ushort_t* __restrict__ xns) {
  long long n = (long long)LNS * 16 * D_;
  long long stride = (long long)gridDim.x * blockDim.x;
  for (long long i = (long long)blockIdx.x * blockDim.x + threadIdx.x; i < n; i += stride) {
    int t = (int)(i / (16 * D_));
    int r = (int)(i % (16 * D_));
    int m = r / D_, d = r % D_;
    float v = (m < 8) ? x[((size_t)m * T_ + (S_ + t)) * D_ + d] : 0.0f;
    xns[i] = f2bf(v);
  }
}

// ------------------------ shared-path GEMM kernels -------------------------
// Each block: 8 waves; each wave: one 16-row tile x four 16-col tiles.

__global__ void k_gemm1_shared(const ushort_t* __restrict__ Xbf,
                               const ushort_t* __restrict__ W1,   // (F,D) N-major
                               ushort_t* __restrict__ H) {
  const int lane = threadIdx.x & 31, wave = threadIdx.x >> 5;
  const int ksel = lane >> 4, nlan = lane & 15;
  const int mtile = blockIdx.x;
  const int b   = mtile / (S_ / 16);
  const int tt0 = (mtile % (S_ / 16)) * 16;
  const ushort_t* Arow = Xbf + ((size_t)b * T_ + tt0 + nlan) * D_;
  const int n0 = blockIdx.y * 512 + wave * 64;

  v8f acc[4];
#pragma unroll
  for (int i = 0; i < 4; ++i)
#pragma unroll
    for (int j = 0; j < 8; ++j) acc[i][j] = 0.0f;

  for (int k0 = 0; k0 < D_; k0 += 32) {
    v16bf a = frag_a(Arow + k0, ksel);
#pragma unroll
    for (int i = 0; i < 4; ++i) {
      v16bf bm = frag_b_nmajor(W1 + (size_t)(n0 + i * 16 + nlan) * D_ + k0, ksel);
      acc[i] = wmma_bf16(a, bm, acc[i]);
    }
  }
  const size_t r0 = (size_t)mtile * 16;
#pragma unroll
  for (int i = 0; i < 4; ++i)
#pragma unroll
    for (int j = 0; j < 8; ++j) {
      int rm = ksel * 8 + j;
      H[(r0 + rm) * F_ + n0 + i * 16 + nlan] = f2bf(gelu_exact(acc[i][j]));
    }
}

__global__ void k_gemm2_shared(const ushort_t* __restrict__ H,    // (B*s, F) bf16
                               const ushort_t* __restrict__ W2,   // (D,F) N-major
                               float* __restrict__ out) {
  const int lane = threadIdx.x & 31, wave = threadIdx.x >> 5;
  const int ksel = lane >> 4, nlan = lane & 15;
  const int mtile = blockIdx.x;
  const int b   = mtile / (S_ / 16);
  const int tt0 = (mtile % (S_ / 16)) * 16;
  const ushort_t* Arow = H + ((size_t)mtile * 16 + nlan) * F_;
  const int n0 = blockIdx.y * 512 + wave * 64;

  v8f acc[4];
#pragma unroll
  for (int i = 0; i < 4; ++i)
#pragma unroll
    for (int j = 0; j < 8; ++j) acc[i][j] = 0.0f;

  for (int k0 = 0; k0 < F_; k0 += 32) {
    v16bf a = frag_a(Arow + k0, ksel);
#pragma unroll
    for (int i = 0; i < 4; ++i) {
      v16bf bm = frag_b_nmajor(W2 + (size_t)(n0 + i * 16 + nlan) * F_ + k0, ksel);
      acc[i] = wmma_bf16(a, bm, acc[i]);
    }
  }
#pragma unroll
  for (int i = 0; i < 4; ++i)
#pragma unroll
    for (int j = 0; j < 8; ++j) {
      int rm = ksel * 8 + j;
      out[((size_t)b * T_ + tt0 + rm) * D_ + n0 + i * 16 + nlan] = acc[i][j];
    }
}

// --------------------- per-timestep (NS) GEMM kernels ----------------------
// One t per blockIdx.x, 256 output cols per block. fp32 weight tiles
// (32 K-rows x 256 cols = 32KB) streamed into LDS by the TDM, double-buffered;
// wave 0 drives the DMA (TENSORcnt is per-wave), barriers publish to the block.

template <int KTOT, int LDN, bool GELU_BF16_OUT>
__global__ void k_ns_gemm_tdm(const ushort_t* __restrict__ Abf,  // [LNS][16][KTOT]
                              const float* __restrict__ Wns,     // [LNS][KTOT][LDN]
                              void* __restrict__ outv) {
  __shared__ float lds[2][32 * 256];
  const int lane = threadIdx.x & 31, wave = threadIdx.x >> 5;
  const int ksel = lane >> 4, nlan = lane & 15;
  const int t = blockIdx.x;
  const int nblk = blockIdx.y * 256;
  const ushort_t* Arow = Abf + ((size_t)t * 16 + nlan) * KTOT;
  const float* Wtile = Wns + (size_t)t * KTOT * LDN + nblk;   // tile col origin
  const int nw = wave * 32;                                   // wave's cols in block

  v8f acc[2];
#pragma unroll
  for (int i = 0; i < 2; ++i)
#pragma unroll
    for (int j = 0; j < 8; ++j) acc[i][j] = 0.0f;

  constexpr int NK = KTOT / 32;
  if (wave == 0)
    tdm_load_2d_f32((unsigned)(uintptr_t)&lds[0][0], Wtile, 256u, 32u, (unsigned)LDN);

  for (int i = 0; i < NK; ++i) {
    if (wave == 0) {
      if (i + 1 < NK) {
        tdm_load_2d_f32((unsigned)(uintptr_t)&lds[(i + 1) & 1][0],
                        Wtile + (size_t)(i + 1) * 32 * LDN, 256u, 32u, (unsigned)LDN);
        __builtin_amdgcn_s_wait_tensorcnt(1);   // tile i has landed
      } else {
        __builtin_amdgcn_s_wait_tensorcnt(0);
      }
    }
    __syncthreads();                            // tile i visible to all waves
    const float* buf = &lds[i & 1][0];
    v16bf a = frag_a(Arow + i * 32, ksel);
#pragma unroll
    for (int f = 0; f < 2; ++f) {
      const int col = nw + f * 16 + nlan;
      v16bf bm;
#pragma unroll
      for (int j = 0; j < 16; ++j)
        bm[j] = (__bf16)buf[(ksel * 16 + j) * 256 + col];
      acc[f] = wmma_bf16(a, bm, acc[f]);
    }
    __syncthreads();                            // done reading before overwrite
  }

#pragma unroll
  for (int f = 0; f < 2; ++f)
#pragma unroll
    for (int j = 0; j < 8; ++j) {
      int rm = ksel * 8 + j;
      int col = nblk + nw + f * 16 + nlan;
      float v = acc[f][j];
      if (GELU_BF16_OUT) {
        ((ushort_t*)outv)[((size_t)t * 16 + rm) * LDN + col] = f2bf(gelu_exact(v));
      } else {
        if (rm < 8)   // rows 8..15 are M-padding (batch has only 8)
          ((float*)outv)[((size_t)rm * T_ + (S_ + t)) * LDN + col] = v;
      }
    }
}

// ------------------------------- launcher ----------------------------------

extern "C" void kernel_launch(void* const* d_in, const int* in_sizes, int n_in,
                              void* d_out, int out_size, void* d_ws, size_t ws_size,
                              hipStream_t stream) {
  (void)in_sizes; (void)n_in; (void)out_size; (void)ws_size;
  const float* x    = (const float*)d_in[0];
  const float* W1S  = (const float*)d_in[1];
  const float* W2S  = (const float*)d_in[2];
  const float* W1NS = (const float*)d_in[3];
  const float* W2NS = (const float*)d_in[4];
  float* out = (float*)d_out;

  // Workspace partition (bf16 elements); total ~113 MB.
  ushort_t* Xbf   = (ushort_t*)d_ws;
  ushort_t* W1Sbf = Xbf   + (size_t)B_ * T_ * D_;
  ushort_t* W2Sbf = W1Sbf + (size_t)F_ * D_;
  ushort_t* Hbf   = W2Sbf + (size_t)D_ * F_;
  ushort_t* Xns   = Hbf   + (size_t)B_ * S_ * F_;
  ushort_t* Hns   = Xns   + (size_t)LNS * 16 * D_;

  // 1) bf16 conversions / packing
  k_f32_to_bf16<<<2048, 256, 0, stream>>>(x,   Xbf,   (long long)B_ * T_ * D_);
  k_f32_to_bf16<<<512,  256, 0, stream>>>(W1S, W1Sbf, (long long)F_ * D_);
  k_f32_to_bf16<<<512,  256, 0, stream>>>(W2S, W2Sbf, (long long)D_ * F_);
  k_build_xns  <<<512,  256, 0, stream>>>(x, Xns);

  // 2) shared path: H = gelu(Xs @ W1S^T);  Y = H @ W2S^T
  k_gemm1_shared<<<dim3(B_ * S_ / 16, F_ / 512), 256, 0, stream>>>(Xbf, W1Sbf, Hbf);
  k_gemm2_shared<<<dim3(B_ * S_ / 16, D_ / 512), 256, 0, stream>>>(Hbf, W2Sbf, out);

  // 3) per-timestep path: TDM-streamed fp32 weights, double-buffered in LDS
  k_ns_gemm_tdm<D_, F_, true ><<<dim3(LNS, F_ / 256), 256, 0, stream>>>(Xns, W1NS, (void*)Hns);
  k_ns_gemm_tdm<F_, D_, false><<<dim3(LNS, D_ / 256), 256, 0, stream>>>(Hns, W2NS, (void*)out);
}
